// Attention_49263274885969
// MI455X (gfx1250) — matrix operand order repaired and checked by reading, not compiled, not run on hardware
//
#include <hip/hip_runtime.h>
#include <hip/hip_bf16.h>

#define C_DIM 1024
#define NIN   2048
#define NTOK  2049
#define B_NUM 8

typedef __bf16 bf16_t;
typedef __attribute__((ext_vector_type(16))) __bf16 v16bf;
typedef __attribute__((ext_vector_type(8)))  float  v8f;

__device__ __forceinline__ float wave_reduce_add(float v) {
#pragma unroll
    for (int m = 16; m > 0; m >>= 1) v += __shfl_xor(v, m, 32);
    return v;
}
__device__ __forceinline__ float wave_reduce_max(float v) {
#pragma unroll
    for (int m = 16; m > 0; m >>= 1) v = fmaxf(v, __shfl_xor(v, m, 32));
    return v;
}

// ---------------- zero init: q0 (1024) and y (32*1024) ----------------
__global__ void k_zero(float* q0, float* y) {
    int t = blockIdx.x * 256 + threadIdx.x;
    if (t < 1024) q0[t] = 0.0f;
    if (t < 32 * 1024) y[t] = 0.0f;
}

// ------- convert weights to bf16 hi/lo (for compensated WMMA) ---------
__global__ void k_convert(const float* __restrict__ Wqkv, const float* __restrict__ Wv,
                          bf16_t* wvs_hi, bf16_t* wvs_lo, bf16_t* wv_hi, bf16_t* wv_lo) {
    int idx = blockIdx.x * 256 + threadIdx.x;      // 0 .. 1M-1
    int k = idx >> 10, j = idx & 1023;
    float f  = Wqkv[(size_t)k * 3072 + 2048 + j];  // V-projection slice of Wqkv
    bf16_t h = (bf16_t)f;
    wvs_hi[idx] = h;
    wvs_lo[idx] = (bf16_t)(f - (float)h);
    float g  = Wv[idx];
    bf16_t h2 = (bf16_t)g;
    wv_hi[idx] = h2;
    wv_lo[idx] = (bf16_t)(g - (float)h2);
}

// ---------------- q0 = tmp @ Wq + bq  (split-K, atomics) --------------
__global__ void k_q0(const float* __restrict__ tmp, const float* __restrict__ Wqkv,
                     const float* __restrict__ bqkv, float* q0) {
    int j  = blockIdx.x * 256 + threadIdx.x;   // output column 0..1023
    int c0 = blockIdx.y * 64;                  // K split
    float acc = 0.0f;
    for (int c = c0; c < c0 + 64; ++c)
        acc += tmp[c] * Wqkv[(size_t)c * 3072 + j];
    if (blockIdx.y == 0) acc += bqkv[j];
    atomicAdd(&q0[j], acc);
}

// --------- u[c][h] = (1/16) * sum_dh Wk[c, h*256+dh] * q0[h*256+dh] ---
__global__ void k_u(const float* __restrict__ Wqkv, const float* __restrict__ q0,
                    float* u4) {
    int lane = threadIdx.x & 31;
    int gw   = blockIdx.x * 8 + (threadIdx.x >> 5);  // 0..4095
    int c = gw >> 2, h = gw & 3;
    const float* wrow = Wqkv + (size_t)c * 3072 + 1024 + h * 256;
    const float* qh   = q0 + h * 256;
    float acc = 0.0f;
#pragma unroll
    for (int i = 0; i < 8; ++i) {
        int dh = lane + 32 * i;
        acc += wrow[dh] * qh[dh];
    }
    acc = wave_reduce_add(acc);
    if (lane == 0) u4[c * 4 + h] = acc * 0.0625f;    // scale = 256^-0.5
}

// ---------------- scores[b][h][n] = x'[b,n,:] . u[:,h] ----------------
__global__ void k_scores(const float* __restrict__ x, const float* __restrict__ tmp,
                         const float* __restrict__ u4, float* scores) {
    int lane = threadIdx.x & 31, w = threadIdx.x >> 5;
    int n = blockIdx.x * 8 + w;
    int b = blockIdx.y;
    if (n >= NTOK) return;
    const float* row = (n == 0) ? tmp : x + ((size_t)b * NIN + (n - 1)) * C_DIM;
    float a0 = 0, a1 = 0, a2 = 0, a3 = 0;
#pragma unroll 4
    for (int i = 0; i < 32; ++i) {
        int c = lane + 32 * i;
        float xv = row[c];
        float4 uu = ((const float4*)u4)[c];
        a0 += xv * uu.x; a1 += xv * uu.y; a2 += xv * uu.z; a3 += xv * uu.w;
    }
    a0 = wave_reduce_add(a0); a1 = wave_reduce_add(a1);
    a2 = wave_reduce_add(a2); a3 = wave_reduce_add(a3);
    if (lane == 0) {
        float* s = scores + (size_t)(b * 4) * NTOK + n;
        s[0] = a0; s[NTOK] = a1; s[2 * NTOK] = a2; s[3 * NTOK] = a3;
    }
}

// ------ softmax over n per (b,h); write transposed att_t[b][n][h] -----
__global__ void k_softmax(const float* __restrict__ scores, float* att_t) {
    int bh = blockIdx.x; int b = bh >> 2, h = bh & 3;
    const float* s = scores + (size_t)bh * NTOK;
    __shared__ float redm[8];
    __shared__ float reds[8];
    int t = threadIdx.x, lane = t & 31, w = t >> 5;
    float mx = -3.4e38f;
    for (int n = t; n < NTOK; n += 256) mx = fmaxf(mx, s[n]);
    mx = wave_reduce_max(mx);
    if (lane == 0) redm[w] = mx;
    __syncthreads();
    mx = redm[0];
#pragma unroll
    for (int i = 1; i < 8; ++i) mx = fmaxf(mx, redm[i]);
    float sum = 0.0f;
    for (int n = t; n < NTOK; n += 256) sum += __expf(s[n] - mx);
    sum = wave_reduce_add(sum);
    if (lane == 0) reds[w] = sum;
    __syncthreads();
    float tot = 0.0f;
#pragma unroll
    for (int i = 0; i < 8; ++i) tot += reds[i];
    float inv = 1.0f / tot;
    for (int n = t; n < NTOK; n += 256)
        att_t[((size_t)b * NTOK + n) * 4 + h] = __expf(s[n] - mx) * inv;
}

// ---- y[b][h][c] = sum_n att[b,h,n] * x'[b,n,c]  (split-n, atomics) ---
__global__ void k_wsum(const float* __restrict__ x, const float* __restrict__ tmp,
                       const float* __restrict__ att_t, float* y) {
    int c  = blockIdx.y * 256 + threadIdx.x;
    int b  = blockIdx.z;
    int n0 = blockIdx.x * 129;
    int n1 = n0 + 129; if (n1 > NTOK) n1 = NTOK;
    float a0 = 0, a1 = 0, a2 = 0, a3 = 0;
    for (int n = n0; n < n1; ++n) {
        const float* row = (n == 0) ? tmp : x + ((size_t)b * NIN + (n - 1)) * C_DIM;
        float xv = row[c];
        float4 at = ((const float4*)att_t)[(size_t)b * NTOK + n];
        a0 += at.x * xv; a1 += at.y * xv; a2 += at.z * xv; a3 += at.w * xv;
    }
    float* yb = y + (size_t)b * 4 * C_DIM + c;
    atomicAdd(yb + 0 * C_DIM, a0);
    atomicAdd(yb + 1 * C_DIM, a1);
    atomicAdd(yb + 2 * C_DIM, a2);
    atomicAdd(yb + 3 * C_DIM, a3);
}

// ----- one 16x16 output tile, K-loop, bf16 hi/lo compensated WMMA -----
// A: f32 rows (mvalid valid rows, others treated as zero), lda stride.
// B: bf16 hi/lo, row-major [K x ldb], columns bcol..bcol+15.
__device__ __forceinline__ void gemm_wave(const float* __restrict__ A, int lda, int mvalid,
                                          const bf16_t* __restrict__ Bhi,
                                          const bf16_t* __restrict__ Blo,
                                          int ldb, int bcol, int K, v8f* accp) {
    int lane = threadIdx.x & 31;
    int m    = lane & 15;
    bool aval = (m < mvalid);
    int mc   = aval ? m : 0;                 // clamped in-bounds row
    int kbA  = (lane < 16) ? 0 : 8;          // ISA 16-bit A 16x32 layout
    int kbB  = (lane < 16) ? 0 : 8;          // same k order as A (table layout)
    int n    = lane & 15;
    const float* arow = A + (size_t)mc * lda;
    float amask = aval ? 1.0f : 0.0f;
    v8f acc = *accp;
    for (int k0 = 0; k0 < K; k0 += 32) {
        v16bf ahi, alo;
#pragma unroll
        for (int e = 0; e < 8; ++e) {
            float f = arow[k0 + kbA + e] * amask;
            bf16_t hh = (bf16_t)f;
            ahi[e] = hh; alo[e] = (bf16_t)(f - (float)hh);
            float f2 = arow[k0 + 16 + kbA + e] * amask;
            bf16_t h2 = (bf16_t)f2;
            ahi[8 + e] = h2; alo[8 + e] = (bf16_t)(f2 - (float)h2);
        }
        v16bf bhi, blo;
#pragma unroll
        for (int e = 0; e < 16; ++e) {
            size_t off = (size_t)(k0 + kbB + (e & 7) + ((e >> 3) << 4)) * ldb + bcol + n;
            bhi[e] = Bhi[off]; blo[e] = Blo[off];
        }
        acc = __builtin_amdgcn_wmma_f32_16x16x32_bf16(false, ahi, false, bhi, (short)0, acc, false, false);
        acc = __builtin_amdgcn_wmma_f32_16x16x32_bf16(false, ahi, false, blo, (short)0, acc, false, false);
        acc = __builtin_amdgcn_wmma_f32_16x16x32_bf16(false, alo, false, bhi, (short)0, acc, false, false);
    }
    *accp = acc;
}

// ------ out_row[b][h*256+j] = y[b,h,:] @ Wv_slice[:, h*256+j] + bias --
__global__ void k_g1(const float* __restrict__ y, const bf16_t* __restrict__ wvs_hi,
                     const bf16_t* __restrict__ wvs_lo, const float* __restrict__ bqkv,
                     float* out_row) {
    int gw = blockIdx.x * 8 + (threadIdx.x >> 5);   // 0..63
    int h  = gw >> 4;
    int nt = gw & 15;
    int bcol = h * 256 + nt * 16;
    v8f acc = {};
    // row m of A is y[(m*4 + h)*1024 + k]  -> base y + h*1024, lda = 4096
    gemm_wave(y + h * 1024, 4096, 8, wvs_hi, wvs_lo, 1024, bcol, 1024, &acc);
    int lane = threadIdx.x & 31;
    int n = lane & 15;
    int mbase = (lane < 16) ? 0 : 8;
#pragma unroll
    for (int i = 0; i < 8; ++i) {
        int m = i + mbase;
        if (m < 8)
            out_row[m * 1024 + bcol + n] = acc[i] + bqkv[2048 + bcol + n];
    }
}

// ---------------- final[b] = out_row[b] @ Wv + bv ---------------------
__global__ void k_g2(const float* __restrict__ out_row, const bf16_t* __restrict__ wv_hi,
                     const bf16_t* __restrict__ wv_lo, const float* __restrict__ bv,
                     float* out) {
    int gw = blockIdx.x * 8 + (threadIdx.x >> 5);   // 0..63 -> N tile
    int bcol = gw * 16;
    v8f acc = {};
    gemm_wave(out_row, 1024, 8, wv_hi, wv_lo, 1024, bcol, 1024, &acc);
    int lane = threadIdx.x & 31;
    int n = lane & 15;
    int mbase = (lane < 16) ? 0 : 8;
#pragma unroll
    for (int i = 0; i < 8; ++i) {
        int m = i + mbase;
        if (m < 8)
            out[m * 1024 + bcol + n] = acc[i] + bv[bcol + n];
    }
}

// ---------------------------- launcher --------------------------------
extern "C" void kernel_launch(void* const* d_in, const int* in_sizes, int n_in,
                              void* d_out, int out_size, void* d_ws, size_t ws_size,
                              hipStream_t stream) {
    const float* x    = (const float*)d_in[0];   // [8,2048,1024]
    const float* tmp  = (const float*)d_in[1];   // [1,1,1024]
    const float* Wqkv = (const float*)d_in[2];   // [1024,3072]
    const float* bqkv = (const float*)d_in[3];   // [3072]
    const float* Wv   = (const float*)d_in[4];   // [1024,1024]
    const float* bv   = (const float*)d_in[5];   // [1024]
    float* out = (float*)d_out;                  // [8,1024]

    char* ws = (char*)d_ws;
    size_t off = 0;
    auto alloc = [&](size_t bytes) { char* p = ws + off; off = (off + bytes + 255) & ~(size_t)255; return p; };
    float*  q0      = (float*)alloc(1024 * 4);
    float*  u4      = (float*)alloc(1024 * 4 * 4);
    float*  scores  = (float*)alloc((size_t)32 * NTOK * 4);
    float*  att_t   = (float*)alloc((size_t)B_NUM * NTOK * 4 * 4);
    float*  y       = (float*)alloc((size_t)32 * 1024 * 4);
    float*  out_row = (float*)alloc((size_t)8 * 1024 * 4);
    bf16_t* wvs_hi  = (bf16_t*)alloc((size_t)1024 * 1024 * 2);
    bf16_t* wvs_lo  = (bf16_t*)alloc((size_t)1024 * 1024 * 2);
    bf16_t* wv_hi   = (bf16_t*)alloc((size_t)1024 * 1024 * 2);
    bf16_t* wv_lo   = (bf16_t*)alloc((size_t)1024 * 1024 * 2);

    k_zero   <<<128, 256, 0, stream>>>(q0, y);
    k_convert<<<4096, 256, 0, stream>>>(Wqkv, Wv, wvs_hi, wvs_lo, wv_hi, wv_lo);
    k_q0     <<<dim3(4, 16), 256, 0, stream>>>(tmp, Wqkv, bqkv, q0);
    k_u      <<<512, 256, 0, stream>>>(Wqkv, q0, u4);
    k_scores <<<dim3(257, 8), 256, 0, stream>>>(x, tmp, u4, scores);
    k_softmax<<<32, 256, 0, stream>>>(scores, att_t);
    k_wsum   <<<dim3(16, 4, 8), 256, 0, stream>>>(x, tmp, att_t, y);
    k_g1     <<<8, 256, 0, stream>>>(y, wvs_hi, wvs_lo, bqkv, out_row);
    k_g2     <<<8, 256, 0, stream>>>(out_row, wv_hi, wv_lo, bv, out);
}
